// FSLPre_43207370997869
// MI455X (gfx1250) — matrix operand order repaired and checked by reading, not compiled, not run on hardware
//
#include <hip/hip_runtime.h>
#include <math.h>

typedef __bf16 bf16;
typedef __attribute__((ext_vector_type(16))) __bf16 v16bf;
typedef __attribute__((ext_vector_type(8)))  float   v8f;

#define N_TOK 784
#define DD    64
#define SS    64
#define BBATCH 128
#define NWAY  16
#define NSHOT 8
#define CBB   512
#define IMG   (DD * N_TOK)   // 50176 elements per image

// ---- gfx1250 async global->LDS path (guarded; sync LDS staging fallback) ----
#if defined(__has_builtin)
#  if __has_builtin(__builtin_amdgcn_global_load_async_to_lds_b128) && \
      __has_builtin(__builtin_amdgcn_s_wait_asynccnt)
#    define HAVE_ASYNC 1
#  endif
#endif
#ifndef HAVE_ASYNC
#  define HAVE_ASYNC 0
#endif

#if HAVE_ASYNC
// Builtin expects: (global int4* src, lds int4* dst, imm offset, imm cpol)
typedef int v4i_ __attribute__((vector_size(16)));
typedef __attribute__((address_space(1))) v4i_ gv4i_t;
typedef __attribute__((address_space(3))) v4i_ lv4i_t;
#endif

static __device__ __forceinline__ float sigmoidf_(float x) {
    return 1.0f / (1.0f + __expf(-x));
}

// ---------------------------------------------------------------------------
// Kernel 1: 1x1 conv GEMM.  xm[d][n] = conv_w[64x512] x support[b][512x784]+b
// WMMA bf16: A = conv_w (M=d, K=c), B = support (K=c, N=token).
// B operand is staged through LDS with double-buffered async b128 loads
// (ASYNCcnt) when available; fragments are then gathered from LDS.
// Writes x fp32 [b][n][d] -> d_out, x_pe fp32 [b][n][d] -> d_out,
// xpe bf16 [b][d][n] -> ws (B-operand layout for the k projection).
// ---------------------------------------------------------------------------
__global__ __launch_bounds__(256) void k_conv(
    const float* __restrict__ support, const float* __restrict__ conv_w,
    const float* __restrict__ conv_b,  const float* __restrict__ pos_emb,
    float* __restrict__ x_out, float* __restrict__ xpe_out,
    bf16* __restrict__ xpe_dn)
{
    __shared__ alignas(16) float stage[8][2][32 * 16];   // 32 KB, per-wave dbl buf
    int wave = threadIdx.x >> 5;
    int lane = threadIdx.x & 31;
    int half = lane >> 4;
    int mrow = lane & 15;
    int ncol = lane & 15;
    int tile = blockIdx.x * 8 + wave;        // 128 images * 196 tiles = 25088
    int b  = tile / 196;
    int t  = tile % 196;
    int tm = t / 49;                          // d tile (0..3)
    int tn = t % 49;                          // token tile (0..48)

    const float* Bm = support + (size_t)b * CBB * N_TOK + tn * 16;

    int srow  = lane >> 2;          // 0..7 (plus i*8): K-row within stage tile
    int scol4 = (lane & 3) * 4;     // float column offset (16B chunk)
    float* buf0 = &stage[wave][0][0];
    float* buf1 = &stage[wave][1][0];

    auto issue_stage = [&](float* dst, int kbase) {
#pragma unroll
        for (int i = 0; i < 4; ++i) {
            int r = i * 8 + srow;                                  // K-row 0..31
            const float* g = Bm + (size_t)(kbase + r) * N_TOK + scol4;
            float* l = dst + r * 16 + scol4;
#if HAVE_ASYNC
            __builtin_amdgcn_global_load_async_to_lds_b128((gv4i_t*)g, (lv4i_t*)l, 0, 0);
#else
            *(float4*)l = *(const float4*)g;
#endif
        }
    };

    v8f acc;
#pragma unroll
    for (int r = 0; r < 8; ++r) acc[r] = conv_b[tm * 16 + r + 8 * half];

    issue_stage(buf0, 0);

    for (int kk = 0; kk < 16; ++kk) {
        int kbase = kk * 32;
        float* cur = (kk & 1) ? buf1 : buf0;
        float* nxt = (kk & 1) ? buf0 : buf1;
        if (kk < 15) issue_stage(nxt, kbase + 32);
#if HAVE_ASYNC
        if (kk < 15) __builtin_amdgcn_s_wait_asynccnt(4);   // cur's 4 ops done
        else         __builtin_amdgcn_s_wait_asynccnt(0);
        __asm__ __volatile__("" ::: "memory");
#endif
        v16bf af, bfv;
#pragma unroll
        for (int v = 0; v < 8; ++v) {
            int ka = 2 * v + half * 16;
            af[2 * v]     = (bf16)conv_w[(tm * 16 + mrow) * CBB + kbase + ka];
            af[2 * v + 1] = (bf16)conv_w[(tm * 16 + mrow) * CBB + kbase + ka + 1];
            int kb = 2 * v + 16 * half;
            bfv[2 * v]     = (bf16)cur[kb * 16 + ncol];
            bfv[2 * v + 1] = (bf16)cur[(kb + 1) * 16 + ncol];
        }
        acc = __builtin_amdgcn_wmma_f32_16x16x32_bf16(false, af, false, bfv,
                                                      (short)0, acc, false, false);
    }

    int n = tn * 16 + ncol;
#pragma unroll
    for (int r = 0; r < 8; ++r) {
        int d = tm * 16 + r + 8 * half;
        float xv = acc[r];
        float xp = xv + pos_emb[d * N_TOK + n];
        size_t idx = (size_t)b * IMG + (size_t)n * DD + d;   // [b][n][d]
        x_out[idx]   = xv;
        xpe_out[idx] = xp;
        xpe_dn[(size_t)b * IMG + (size_t)d * N_TOK + n] = (bf16)xp;   // [b][d][n]
    }
}

// ---------------------------------------------------------------------------
// Kernel 2: k projection. k[d'][n] = to_k_w[64x64] x xpe[b][64x784], * D^-0.5
// Stored bf16 K-pair swizzled: elem(d, n) at (d>>1)*1568 + 2n + (d&1),
// so dots-GEMM B fragments load (k,n),(k+1,n) as one contiguous b32.
// ---------------------------------------------------------------------------
__global__ __launch_bounds__(256) void k_proj(
    const float* __restrict__ to_k_w, const bf16* __restrict__ xpe_dn,
    bf16* __restrict__ kbf)
{
    int wave = threadIdx.x >> 5;
    int lane = threadIdx.x & 31;
    int half = lane >> 4;
    int mrow = lane & 15;
    int ncol = lane & 15;
    int tile = blockIdx.x * 8 + wave;        // 25088
    int b  = tile / 196;
    int t  = tile % 196;
    int tm = t / 49;
    int tn = t % 49;

    const bf16* Bm = xpe_dn + (size_t)b * IMG;
    v8f acc = {};

    for (int kk = 0; kk < 2; ++kk) {
        int kbase = kk * 32;
        v16bf af, bfv;
#pragma unroll
        for (int v = 0; v < 8; ++v) {
            int ka = 2 * v + half * 16;
            af[2 * v]     = (bf16)to_k_w[(tm * 16 + mrow) * DD + kbase + ka];
            af[2 * v + 1] = (bf16)to_k_w[(tm * 16 + mrow) * DD + kbase + ka + 1];
            int kb = 2 * v + 16 * half;
            bfv[2 * v]     = Bm[(size_t)(kbase + kb) * N_TOK + tn * 16 + ncol];
            bfv[2 * v + 1] = Bm[(size_t)(kbase + kb + 1) * N_TOK + tn * 16 + ncol];
        }
        acc = __builtin_amdgcn_wmma_f32_16x16x32_bf16(false, af, false, bfv,
                                                      (short)0, acc, false, false);
    }

    int n = tn * 16 + ncol;
#pragma unroll
    for (int r = 0; r < 8; ++r) {
        int d = tm * 16 + r + 8 * half;
        // K-pair interleaved swizzle, scale D^-0.5 folded in
        kbf[(size_t)b * IMG + (size_t)(d >> 1) * (2 * N_TOK) + 2 * n + (d & 1)]
            = (bf16)(acc[r] * 0.125f);
    }
}

// ---------------------------------------------------------------------------
// Kernel 3: broadcast init_slots into per-image slot state; zero scores.
// ---------------------------------------------------------------------------
__global__ void k_init(const float* __restrict__ init_slots,
                       float* __restrict__ slots, float* __restrict__ scores)
{
    int gid = blockIdx.x * blockDim.x + threadIdx.x;       // 524288 total
    if (gid < BBATCH * SS * DD) slots[gid] = init_slots[gid % (SS * DD)];
    if (gid < NWAY * SS) scores[gid] = 0.0f;
}

// ---------------------------------------------------------------------------
// Kernel 4a: dots[b][s][n] = slots[b][s][:] . k[b][:][n]   (k already scaled)
// B fragments come from the K-pair swizzled k buffer: one b32 per VGPR.
// ---------------------------------------------------------------------------
__global__ __launch_bounds__(256) void k_dots(
    const float* __restrict__ slots, const bf16* __restrict__ kbf,
    float* __restrict__ dots)
{
    int wave = threadIdx.x >> 5;
    int lane = threadIdx.x & 31;
    int half = lane >> 4;
    int mrow = lane & 15;
    int ncol = lane & 15;
    int tile = blockIdx.x * 8 + wave;        // 25088
    int b  = tile / 196;
    int t  = tile % 196;
    int ts = t / 49;                          // slot tile
    int tn = t % 49;                          // token tile

    const float* Am = slots + (size_t)b * SS * DD;
    const bf16*  Bm = kbf   + (size_t)b * IMG;
    int n = tn * 16 + ncol;
    v8f acc = {};

    for (int kk = 0; kk < 2; ++kk) {
        int kbase = kk * 32;
        v16bf af, bfv;
#pragma unroll
        for (int v = 0; v < 8; ++v) {
            int ka = 2 * v + half * 16;
            af[2 * v]     = (bf16)Am[(ts * 16 + mrow) * DD + kbase + ka];
            af[2 * v + 1] = (bf16)Am[(ts * 16 + mrow) * DD + kbase + ka + 1];
            int kb = kbase + 2 * v + 16 * half;          // even
            const bf16* p = Bm + (size_t)(kb >> 1) * (2 * N_TOK) + 2 * n;
            bfv[2 * v]     = p[0];
            bfv[2 * v + 1] = p[1];
        }
        acc = __builtin_amdgcn_wmma_f32_16x16x32_bf16(false, af, false, bfv,
                                                      (short)0, acc, false, false);
    }

#pragma unroll
    for (int r = 0; r < 8; ++r) {
        int s = ts * 16 + r + 8 * half;
        dots[(size_t)b * IMG + (size_t)s * N_TOK + n] = acc[r];
    }
}

// ---------------------------------------------------------------------------
// Kernel 4b: per image: rowsum/total renorm + sigmoid (attn written in place).
// On last iteration also accumulate per-(way, slot) attention mass.
// ---------------------------------------------------------------------------
__global__ __launch_bounds__(256) void k_norm(
    float* __restrict__ dots, float* __restrict__ scores, int last)
{
    __shared__ float rowsum[SS];
    __shared__ float attsum[SS];
    __shared__ float totalsh;
    int b = blockIdx.x;
    int t = threadIdx.x;
    size_t base = (size_t)b * IMG;

    if (t < SS) { rowsum[t] = 0.0f; attsum[t] = 0.0f; }
    __syncthreads();

    for (int i = t; i < IMG; i += 256)
        atomicAdd(&rowsum[i / N_TOK], dots[base + i]);
    __syncthreads();

    if (t == 0) {
        float tot = 0.0f;
        for (int s = 0; s < SS; ++s) tot += rowsum[s];
        totalsh = tot;
    }
    __syncthreads();

    float tot = totalsh;
    for (int i = t; i < IMG; i += 256) {
        int s = i / N_TOK;
        float v = dots[base + i] / rowsum[s] * tot;
        float a = sigmoidf_(v);
        dots[base + i] = a;
        if (last) atomicAdd(&attsum[s], a);
    }
    __syncthreads();

    if (last && t < SS)
        atomicAdd(&scores[(b >> 3) * SS + t], attsum[t]);
}

// ---------------------------------------------------------------------------
// Kernel 4c: updates[b][s][d] = (1/64) * attn[b][s][:784] . x[b][:784][d]
// K=784 handled as 25 steps of 32 with zero-fill for K>=784.
// ---------------------------------------------------------------------------
__global__ __launch_bounds__(256) void k_upd(
    const float* __restrict__ attn, const float* __restrict__ x_f32,
    float* __restrict__ updates)
{
    int wave = threadIdx.x >> 5;
    int lane = threadIdx.x & 31;
    int half = lane >> 4;
    int mrow = lane & 15;
    int ncol = lane & 15;
    int tile = blockIdx.x * 8 + wave;        // 128 * 16 = 2048
    int b  = tile / 16;
    int t  = tile % 16;
    int ts = t / 4;                           // slot tile
    int td = t % 4;                           // d tile

    const float* Am = attn  + (size_t)b * IMG;     // [s][n]
    const float* Bm = x_f32 + (size_t)b * IMG;     // [n][d]
    v8f acc = {};

    for (int kk = 0; kk < 25; ++kk) {
        int kbase = kk * 32;
        v16bf af, bfv;
#pragma unroll
        for (int v = 0; v < 8; ++v) {
            int ka = kbase + 2 * v + half * 16;
            af[2 * v]     = (bf16)((ka     < N_TOK) ? Am[(ts * 16 + mrow) * N_TOK + ka]     : 0.0f);
            af[2 * v + 1] = (bf16)((ka + 1 < N_TOK) ? Am[(ts * 16 + mrow) * N_TOK + ka + 1] : 0.0f);
            int kb = kbase + 2 * v + 16 * half;
            bfv[2 * v]     = (bf16)((kb     < N_TOK) ? Bm[(size_t)kb * DD       + td * 16 + ncol] : 0.0f);
            bfv[2 * v + 1] = (bf16)((kb + 1 < N_TOK) ? Bm[(size_t)(kb + 1) * DD + td * 16 + ncol] : 0.0f);
        }
        acc = __builtin_amdgcn_wmma_f32_16x16x32_bf16(false, af, false, bfv,
                                                      (short)0, acc, false, false);
    }

    int d = td * 16 + ncol;
#pragma unroll
    for (int r = 0; r < 8; ++r) {
        int s = ts * 16 + r + 8 * half;
        updates[(size_t)b * SS * DD + s * DD + d] = acc[r] * (1.0f / 64.0f);
    }
}

// ---------------------------------------------------------------------------
// Kernel 4d: GRU step per (b, slot) row. One 64-thread block per row.
// ---------------------------------------------------------------------------
__global__ __launch_bounds__(64) void k_gru(
    const float* __restrict__ updates, float* __restrict__ slots,
    const float* __restrict__ w_ih, const float* __restrict__ w_hh,
    const float* __restrict__ b_ih, const float* __restrict__ b_hh)
{
    __shared__ float xv[DD];
    __shared__ float hv[DD];
    int row = blockIdx.x;                     // 0..8191
    int j = threadIdx.x;
    xv[j] = updates[(size_t)row * DD + j];
    hv[j] = slots[(size_t)row * DD + j];
    __syncthreads();

    float ir = b_ih[j], iz = b_ih[DD + j], in_ = b_ih[2 * DD + j];
    float hr = b_hh[j], hz = b_hh[DD + j], hn  = b_hh[2 * DD + j];
    for (int d = 0; d < DD; ++d) {
        float xd = xv[d], hd = hv[d];
        ir  += w_ih[j * DD + d] * xd;
        iz  += w_ih[(DD + j) * DD + d] * xd;
        in_ += w_ih[(2 * DD + j) * DD + d] * xd;
        hr  += w_hh[j * DD + d] * hd;
        hz  += w_hh[(DD + j) * DD + d] * hd;
        hn  += w_hh[(2 * DD + j) * DD + d] * hd;
    }
    float r = sigmoidf_(ir + hr);
    float z = sigmoidf_(iz + hz);
    float n = tanhf(in_ + r * hn);
    slots[(size_t)row * DD + j] = (1.0f - z) * n + z * hv[j];
}

// ---------------------------------------------------------------------------
// Kernel 5: per-way argmax -> created slot, plus avg_feature over shots & d.
// ---------------------------------------------------------------------------
__global__ __launch_bounds__(256) void k_final(
    const float* __restrict__ scores, const float* __restrict__ init_slots,
    const float* __restrict__ x_f32, float* __restrict__ out_head)
{
    __shared__ int best;
    int way = blockIdx.x;
    int t = threadIdx.x;
    if (t == 0) {
        float bm = -3.4e38f; int bi = 0;
        for (int s = 0; s < SS; ++s) {
            float v = scores[way * SS + s];
            if (v > bm) { bm = v; bi = s; }
        }
        best = bi;
    }
    __syncthreads();
    if (t < DD) out_head[way * (DD + N_TOK) + t] = init_slots[best * DD + t];

    for (int n = t; n < N_TOK; n += 256) {
        float sum = 0.0f;
        for (int shot = 0; shot < NSHOT; ++shot) {
            const float* xp = x_f32 + ((size_t)(way * NSHOT + shot) * N_TOK + n) * DD;
            for (int d = 0; d < DD; ++d) sum += xp[d];
        }
        out_head[way * (DD + N_TOK) + DD + n] = sum * (1.0f / 512.0f);
    }
}

// ---------------------------------------------------------------------------
extern "C" void kernel_launch(void* const* d_in, const int* in_sizes, int n_in,
                              void* d_out, int out_size, void* d_ws, size_t ws_size,
                              hipStream_t stream)
{
    const float* support    = (const float*)d_in[0];
    const float* conv_w     = (const float*)d_in[1];
    const float* conv_b     = (const float*)d_in[2];
    const float* pos_emb    = (const float*)d_in[3];
    const float* to_k_w     = (const float*)d_in[4];
    const float* gru_w_ih   = (const float*)d_in[5];
    const float* gru_w_hh   = (const float*)d_in[6];
    const float* gru_b_ih   = (const float*)d_in[7];
    const float* gru_b_hh   = (const float*)d_in[8];
    const float* init_slots = (const float*)d_in[9];
    // d_in[10] = split_index: unused by the reference computation.

    // d_out = [out 16x848][x_pe 128x784x64][x 128x784x64]
    float* out_head = (float*)d_out;
    float* xpe_out  = out_head + NWAY * (DD + N_TOK);
    float* x_out    = xpe_out + (size_t)BBATCH * IMG;

    // workspace layout (bytes), ~56 MB total
    char* wsb = (char*)d_ws;
    const size_t SZ_BF_IMG = (size_t)BBATCH * IMG * sizeof(bf16);   // 12,845,056
    bf16*  xpe_dn  = (bf16*)(wsb);
    bf16*  kbf     = (bf16*)(wsb + SZ_BF_IMG);
    float* dots    = (float*)(wsb + 2 * SZ_BF_IMG);
    float* slots   = (float*)(wsb + 2 * SZ_BF_IMG + (size_t)BBATCH * IMG * 4);
    float* updates = slots + (size_t)BBATCH * SS * DD;
    float* scores  = updates + (size_t)BBATCH * SS * DD;

    k_conv<<<3136, 256, 0, stream>>>(support, conv_w, conv_b, pos_emb,
                                     x_out, xpe_out, xpe_dn);
    k_proj<<<3136, 256, 0, stream>>>(to_k_w, xpe_dn, kbf);
    k_init<<<2048, 256, 0, stream>>>(init_slots, slots, scores);

    for (int it = 0; it < 3; ++it) {
        k_dots<<<3136, 256, 0, stream>>>(slots, kbf, dots);
        k_norm<<<128, 256, 0, stream>>>(dots, scores, (it == 2) ? 1 : 0);
        if (it < 2) {   // final attn is produced before the last GRU update
            k_upd<<<256, 256, 0, stream>>>(dots, x_out, updates);
            k_gru<<<8192, 64, 0, stream>>>(updates, slots,
                                           gru_w_ih, gru_w_hh, gru_b_ih, gru_b_hh);
        }
    }

    k_final<<<16, 256, 0, stream>>>(scores, init_slots, x_out, out_head);
}
